// MoE_42975442763861
// MI455X (gfx1250) — matrix operand order repaired and checked by reading, not compiled, not run on hardware
//
#include <hip/hip_runtime.h>
#include <math.h>

typedef __attribute__((ext_vector_type(16))) _Float16 v16h;
typedef __attribute__((ext_vector_type(8)))  float    v8f;

#define H_DIM  2048
#define F_DIM  8192
#define E_NUM  8
#define TM     32            // tokens per workgroup (2 m-subtiles of 16)
#define FC     512           // F-chunk resident in LDS
#define NSPLIT 2             // output H split across blockIdx.z
#define HSPAN  (H_DIM / NSPLIT)
#define HPAD   8             // LDS row padding (halves) -> row stride 16B-aligned

__device__ __forceinline__ float gelu_erf(float x) {
    return 0.5f * x * (1.0f + erff(x * 0.70710678118654752f));
}

// ---------------- gate: logits -> top2 -> softmax -> alpha-folded weights ----------------
__global__ __launch_bounds__(64) void moe_gate(const float* __restrict__ X,
                                               const float* __restrict__ GW,
                                               const float* __restrict__ GB,
                                               const float* __restrict__ AL,
                                               int2* __restrict__ t2e,
                                               float2* __restrict__ t2w, int T) {
    __shared__ float sacc[E_NUM][64];
    __shared__ float slog[E_NUM];
    const int t = blockIdx.x;
    const int tid = threadIdx.x;

    float a[E_NUM];
#pragma unroll
    for (int e = 0; e < E_NUM; ++e) a[e] = 0.0f;

    const float* xr = X + (size_t)t * H_DIM;
    for (int k = tid; k < H_DIM; k += 64) {
        const float xv = xr[k];
        const float* g = GW + (size_t)k * E_NUM;
#pragma unroll
        for (int e = 0; e < E_NUM; ++e) a[e] = fmaf(xv, g[e], a[e]);
    }
#pragma unroll
    for (int e = 0; e < E_NUM; ++e) sacc[e][tid] = a[e];
    __syncthreads();

    if (tid < E_NUM) {
        float s = 0.0f;
#pragma unroll
        for (int i = 0; i < 64; ++i) s += sacc[tid][i];
        slog[tid] = s + GB[tid];
    }
    __syncthreads();

    if (tid == 0) {
        int i1 = 0;
#pragma unroll
        for (int e = 1; e < E_NUM; ++e) if (slog[e] > slog[i1]) i1 = e;
        int i2 = (i1 == 0) ? 1 : 0;
#pragma unroll
        for (int e = 0; e < E_NUM; ++e) if (e != i1 && e != i2 && slog[e] > slog[i2]) i2 = e;
        const float e2 = __expf(slog[i2] - slog[i1]);
        const float d  = 1.0f + e2;
        int2 ee; ee.x = i1; ee.y = i2;
        float2 ww; ww.x = (1.0f / d) * AL[i1]; ww.y = (e2 / d) * AL[i2];
        t2e[t] = ee;
        t2w[t] = ww;
    }
}

// ---------------- routing ----------------
__global__ void route_init(int* __restrict__ cnt) {
    if (threadIdx.x < E_NUM) cnt[threadIdx.x] = 0;
}

__global__ void route_scatter(const int2* __restrict__ t2e, const float2* __restrict__ t2w,
                              int* __restrict__ cnt, int* __restrict__ list,
                              float* __restrict__ wl, int T) {
    const int t = blockIdx.x * blockDim.x + threadIdx.x;
    if (t >= T) return;
    const int2 ee = t2e[t];
    const float2 ww = t2w[t];
    int p0 = atomicAdd(&cnt[ee.x], 1);
    list[(size_t)ee.x * T + p0] = t;  wl[(size_t)ee.x * T + p0] = ww.x;
    int p1 = atomicAdd(&cnt[ee.y], 1);
    list[(size_t)ee.y * T + p1] = t;  wl[(size_t)ee.y * T + p1] = ww.y;
}

// ---------------- out init: out[t,n] = sum_k w_k * b2[e_k, n] ----------------
__global__ __launch_bounds__(256) void out_init(const float* __restrict__ B2,
                                                const int2* __restrict__ t2e,
                                                const float2* __restrict__ t2w,
                                                float* __restrict__ OUT) {
    const size_t i = (size_t)blockIdx.x * 256 + threadIdx.x;
    const int t = (int)(i >> 11);          // /H_DIM
    const int n = (int)(i & (H_DIM - 1));
    const int2 ee = t2e[t];
    const float2 ww = t2w[t];
    OUT[i] = ww.x * B2[(size_t)ee.x * H_DIM + n] + ww.y * B2[(size_t)ee.y * H_DIM + n];
}

// ---------------- fused sparse expert FFN (f16 WMMA, f32 accum) ----------------
__global__ __launch_bounds__(256, 1) void moe_ffn(const float* __restrict__ X,
                                                  const float* __restrict__ W1,
                                                  const float* __restrict__ B1,
                                                  const float* __restrict__ W2,
                                                  const int* __restrict__ list,
                                                  const float* __restrict__ wl,
                                                  const int* __restrict__ cnt,
                                                  float* __restrict__ OUT, int T) {
    const int e    = blockIdx.y;
    const int tile = blockIdx.x;
    const int ne   = cnt[e];
    if (tile * TM >= ne) return;   // uniform early-exit for empty tiles

    __shared__ _Float16 hbuf[TM][FC + HPAD];   // gelu(fc1) chunk, f16
    __shared__ int   stok[TM];
    __shared__ float swt[TM];

    const int tid  = threadIdx.x;
    const int wave = tid >> 5;
    const int lane = tid & 31;
    const int row  = lane & 15;    // A-frag M row / B-frag N col
    const int hi   = lane >> 4;    // lane half selects K sub-ranges
    const int nbase = blockIdx.z * HSPAN;

    if (tid < TM) {
        const int slot = tile * TM + tid;
        const int cs   = (slot < ne) ? slot : (ne - 1);
        stok[tid] = list[(size_t)e * T + cs];
        swt[tid]  = (slot < ne) ? wl[(size_t)e * T + cs] : 0.0f;   // padded rows add 0
    }
    __syncthreads();

    const float* W1e = W1 + (size_t)e * H_DIM * F_DIM;
    const float* W2e = W2 + (size_t)e * F_DIM * H_DIM;
    const float* B1e = B1 + (size_t)e * F_DIM;

    const float* xr0 = X + (size_t)stok[row] * H_DIM;        // m-subtile 0 rows
    const float* xr1 = X + (size_t)stok[16 + row] * H_DIM;   // m-subtile 1 rows

    v8f yacc[16] = {};   // [j:8 n-tiles][m:2]

    for (int c = 0; c < F_DIM / FC; ++c) {
        const int fbase = c * FC;

        // ---- fc1: h[32, FC] = gelu(x_tile @ W1[:, fbase:fbase+FC] + b1) ----
        v8f acc[8] = {};   // [j:4][m:2]
        for (int kb = 0; kb < H_DIM; kb += 32) {
            v16h a0, a1;
            {
                const float4* p = (const float4*)(xr0 + kb + hi * 8);
                const float4* q = (const float4*)(xr0 + kb + 16 + hi * 8);
                const float4 f0 = p[0], f1 = p[1], g0 = q[0], g1 = q[1];
                a0[0] = (_Float16)f0.x; a0[1] = (_Float16)f0.y; a0[2] = (_Float16)f0.z; a0[3] = (_Float16)f0.w;
                a0[4] = (_Float16)f1.x; a0[5] = (_Float16)f1.y; a0[6] = (_Float16)f1.z; a0[7] = (_Float16)f1.w;
                a0[8] = (_Float16)g0.x; a0[9] = (_Float16)g0.y; a0[10] = (_Float16)g0.z; a0[11] = (_Float16)g0.w;
                a0[12] = (_Float16)g1.x; a0[13] = (_Float16)g1.y; a0[14] = (_Float16)g1.z; a0[15] = (_Float16)g1.w;
            }
            {
                const float4* p = (const float4*)(xr1 + kb + hi * 8);
                const float4* q = (const float4*)(xr1 + kb + 16 + hi * 8);
                const float4 f0 = p[0], f1 = p[1], g0 = q[0], g1 = q[1];
                a1[0] = (_Float16)f0.x; a1[1] = (_Float16)f0.y; a1[2] = (_Float16)f0.z; a1[3] = (_Float16)f0.w;
                a1[4] = (_Float16)f1.x; a1[5] = (_Float16)f1.y; a1[6] = (_Float16)f1.z; a1[7] = (_Float16)f1.w;
                a1[8] = (_Float16)g0.x; a1[9] = (_Float16)g0.y; a1[10] = (_Float16)g0.z; a1[11] = (_Float16)g0.w;
                a1[12] = (_Float16)g1.x; a1[13] = (_Float16)g1.y; a1[14] = (_Float16)g1.z; a1[15] = (_Float16)g1.w;
            }
#pragma unroll
            for (int j = 0; j < 4; ++j) {
                const int ncol = fbase + ((j << 3) + wave) * 16 + row;
                const float* bp = W1e + (size_t)(kb + hi * 8) * F_DIM + ncol;
                v16h b;
#pragma unroll
                for (int i = 0; i < 16; ++i) b[i] = (_Float16)bp[(size_t)(i < 8 ? i : i + 8) * F_DIM];
                acc[j * 2 + 0] = __builtin_amdgcn_wmma_f32_16x16x32_f16(
                    false, a0, false, b, (short)0, acc[j * 2 + 0], false, false);
                acc[j * 2 + 1] = __builtin_amdgcn_wmma_f32_16x16x32_f16(
                    false, a1, false, b, (short)0, acc[j * 2 + 1], false, false);
            }
        }
        // bias + exact GELU -> LDS f16
#pragma unroll
        for (int j = 0; j < 4; ++j) {
            const int n0 = ((j << 3) + wave) * 16 + row;   // chunk-local col
            const float b1v = B1e[fbase + n0];
#pragma unroll
            for (int m = 0; m < 2; ++m)
#pragma unroll
                for (int r = 0; r < 8; ++r) {
                    const int mm = m * 16 + r + hi * 8;
                    hbuf[mm][n0] = (_Float16)gelu_erf(acc[j * 2 + m][r] + b1v);
                }
        }
        __syncthreads();

        // ---- fc2 partial: y += h_chunk @ W2[fbase:fbase+FC, nbase:nbase+HSPAN] ----
        for (int kb = 0; kb < FC; kb += 32) {
            v16h a0, a1;
#pragma unroll
            for (int i = 0; i < 8; ++i) {
                a0[i]     = hbuf[row][kb + hi * 8 + i];
                a0[8 + i] = hbuf[row][kb + 16 + hi * 8 + i];
                a1[i]     = hbuf[16 + row][kb + hi * 8 + i];
                a1[8 + i] = hbuf[16 + row][kb + 16 + hi * 8 + i];
            }
#pragma unroll
            for (int j = 0; j < 8; ++j) {
                const int ncol = nbase + ((j << 3) + wave) * 16 + row;
                const float* bp = W2e + (size_t)(fbase + kb + hi * 8) * H_DIM + ncol;
                v16h b;
#pragma unroll
                for (int i = 0; i < 16; ++i) b[i] = (_Float16)bp[(size_t)(i < 8 ? i : i + 8) * H_DIM];
                yacc[j * 2 + 0] = __builtin_amdgcn_wmma_f32_16x16x32_f16(
                    false, a0, false, b, (short)0, yacc[j * 2 + 0], false, false);
                yacc[j * 2 + 1] = __builtin_amdgcn_wmma_f32_16x16x32_f16(
                    false, a1, false, b, (short)0, yacc[j * 2 + 1], false, false);
            }
        }
        __syncthreads();
    }

    // ---- weighted scatter-add into out ----
#pragma unroll
    for (int j = 0; j < 8; ++j) {
        const int n = nbase + ((j << 3) + wave) * 16 + row;
#pragma unroll
        for (int m = 0; m < 2; ++m)
#pragma unroll
            for (int r = 0; r < 8; ++r) {
                const int mm = m * 16 + r + hi * 8;
                atomicAdd(&OUT[(size_t)stok[mm] * H_DIM + n], swt[mm] * yacc[j * 2 + m][r]);
            }
    }
}

extern "C" void kernel_launch(void* const* d_in, const int* in_sizes, int n_in,
                              void* d_out, int out_size, void* d_ws, size_t ws_size,
                              hipStream_t stream) {
    const float* X  = (const float*)d_in[0];   // hidden_states [2,2048,H]
    const float* GW = (const float*)d_in[1];   // gate_w [H,E]
    const float* GB = (const float*)d_in[2];   // gate_b [E]
    const float* W1 = (const float*)d_in[3];   // fc1_w [E,H,F]
    const float* B1 = (const float*)d_in[4];   // fc1_b [E,F]
    const float* W2 = (const float*)d_in[5];   // fc2_w [E,F,H]
    const float* B2 = (const float*)d_in[6];   // fc2_b [E,H]
    const float* AL = (const float*)d_in[7];   // alpha [E]
    float* OUT = (float*)d_out;

    const int T = in_sizes[0] / H_DIM;         // 4096 tokens

    // workspace layout (~320 KB total)
    char* w = (char*)d_ws;
    int*    cnt  = (int*)w;                                     // [E]
    int2*   t2e  = (int2*)(w + 256);                            // [T]
    float2* t2w  = (float2*)(w + 256 + (size_t)T * 8);          // [T]
    int*    list = (int*)(w + 256 + (size_t)T * 16);            // [E,T]
    float*  wl   = (float*)(w + 256 + (size_t)T * 16 + (size_t)E_NUM * T * 4); // [E,T]

    route_init<<<1, 64, 0, stream>>>(cnt);
    moe_gate<<<T, 64, 0, stream>>>(X, GW, GB, AL, t2e, t2w, T);
    route_scatter<<<(T + 255) / 256, 256, 0, stream>>>(t2e, t2w, cnt, list, wl, T);
    out_init<<<(T * H_DIM) / 256, 256, 0, stream>>>(B2, t2e, t2w, OUT);
    moe_ffn<<<dim3(T / TM, E_NUM, NSPLIT), 256, 0, stream>>>(X, W1, B1, W2, list, wl, cnt, OUT, T);
}
